// SimpleAttention_47983374631299
// MI455X (gfx1250) — matrix operand order repaired and checked by reading, not compiled, not run on hardware
//
#include <hip/hip_runtime.h>
#include <stdint.h>

// ---------------------------------------------------------------------------
// SimpleAttention on MI455X (gfx1250, wave32, WMMA + async-to-LDS).
// All heavy math uses v_wmma_f32_16x16x32_bf16 (f32 accumulate).
// GEMMs stage tiles in LDS with global_load_async_to_lds_b128 (ASYNCcnt),
// double-buffered to overlap copy and compute.
// ---------------------------------------------------------------------------

#define BSZ    4
#define SEQ    2000
#define HID    1024
#define NHEAD  16
#define HDIM   64
#define MROWS  (BSZ * SEQ)          // 8000
#define NELEM  (MROWS * HID)        // 8,192,000

typedef __attribute__((ext_vector_type(16))) __bf16       bf16x16;
typedef __attribute__((ext_vector_type(8)))  float        f32x8;
typedef __attribute__((ext_vector_type(4)))  unsigned int u32x4;

union Frag { bf16x16 v; u32x4 q[2]; };   // trivially-constructible 32B fragment

__device__ __forceinline__ uint16_t f2b(float f) {
  union { __bf16 b; uint16_t u; } c;
  c.b = (__bf16)f;
  return c.u;
}

// 16B global -> LDS async copy (per lane). Tracked by ASYNCcnt.
__device__ __forceinline__ void async_b128(uint32_t lds_addr, const uint16_t* g) {
  asm volatile("global_load_async_to_lds_b128 %0, %1, off"
               :: "v"(lds_addr), "v"(g) : "memory");
}
__device__ __forceinline__ void wait_async0() {
  asm volatile("s_wait_asynccnt 0x0" ::: "memory");
}

// ---------------------------------------------------------------------------
// h = hidden_states + pos_emb (broadcast over batch), f32 -> bf16
// ---------------------------------------------------------------------------
__global__ void k_addpos(const float* __restrict__ hs,
                         const float* __restrict__ pos,
                         uint16_t* __restrict__ out) {
  int i = blockIdx.x * blockDim.x + threadIdx.x;
  if (i >= NELEM) return;
  int col = i & (HID - 1);
  int s   = (i >> 10) % SEQ;
  out[i] = f2b(hs[i] + pos[s * HID + col]);
}

// f32 -> bf16 bulk convert (weights)
__global__ void k_f2bf(const float* __restrict__ in,
                       uint16_t* __restrict__ out, int n) {
  int i = blockIdx.x * blockDim.x + threadIdx.x;
  if (i >= n) return;
  out[i] = f2b(in[i]);
}

// V (b,s,h,d) -> Vt (b,h,d,s)  so PV B-fragments read contiguous key runs.
__global__ void k_vtrans(const uint16_t* __restrict__ v,
                         uint16_t* __restrict__ vt) {
  int i = blockIdx.x * blockDim.x + threadIdx.x;
  if (i >= NELEM) return;
  int s  = i % SEQ;
  int r  = i / SEQ;
  int d  = r & (HDIM - 1);
  int bh = r >> 6;
  int h  = bh & (NHEAD - 1);
  int b  = bh >> 4;
  vt[i] = v[((size_t)(b * SEQ + s) << 10) + (h << 6) + d];
}

// ---------------------------------------------------------------------------
// C(MxN) = A(MxK) * B(NxK)^T, bf16 in, f32 accumulate.
// Block = 256 threads = 8 waves. Block tile 128x128, K-step 32, LDS
// double-buffered via global_load_async_to_lds_b128. Wave tile 32x64
// (2 M-frags x 4 N-frags = 8 WMMA per K-step, fed by ds_load_b128).
// Fragments use contiguous k order (k = 16*(lane/16) + s), same for A and B.
// ---------------------------------------------------------------------------
template <bool OUTF>
__global__ __launch_bounds__(256) void k_gemm(const uint16_t* __restrict__ A,
                                              const uint16_t* __restrict__ Bm,
                                              float* __restrict__ Cf,
                                              uint16_t* __restrict__ Cb,
                                              int M, int N, int K) {
  __shared__ uint16_t sA[2][128 * 32];   // 8 KB per buffer
  __shared__ uint16_t sB[2][128 * 32];   // 8 KB per buffer  (32 KB total)

  const int t     = threadIdx.x;
  const int lane  = t & 31;
  const int wid   = t >> 5;
  const int lid   = lane & 15;
  const int lhi   = (lane >> 4) & 1;
  const int m0blk = blockIdx.y * 128;
  const int n0blk = blockIdx.x * 128;
  const int waveM = wid >> 1;            // 0..3  -> 32-row strip
  const int waveN = wid & 1;             // 0..1  -> 64-col strip

  // Stage a 128x32 A tile and 128x32 B tile into LDS buffer `buf`.
  auto stage = [&](int buf, int k0) {
#pragma unroll
    for (int i = 0; i < 2; ++i) {
      int segid = t + 256 * i;           // 0..511 : (row, 16B-segment)
      int row   = segid >> 2;
      int seg   = segid & 3;
      int rA    = m0blk + row; if (rA > M - 1) rA = M - 1;   // clamp tail rows
      async_b128((uint32_t)(size_t)&sA[buf][row * 32 + seg * 8],
                 A + (size_t)rA * K + k0 + seg * 8);
      async_b128((uint32_t)(size_t)&sB[buf][row * 32 + seg * 8],
                 Bm + (size_t)(n0blk + row) * K + k0 + seg * 8);
    }
  };

  f32x8 acc[2][4] = {};
  const int kA0 = lhi * 16;   // A: lane-half selects 16-elem K group
  const int kB0 = lhi * 16;   // B: lane-half selects 16-elem K group

  stage(0, 0);
  int cur = 0;
  for (int k0 = 0; k0 < K; k0 += 32) {
    wait_async0();
    __syncthreads();
    if (k0 + 32 < K) stage(cur ^ 1, k0 + 32);

    Frag a[2], b[4];
#pragma unroll
    for (int mi = 0; mi < 2; ++mi) {
      const uint16_t* p = &sA[cur][(waveM * 32 + mi * 16 + lid) * 32 + kA0];
      a[mi].q[0] = *(const u32x4*)p;
      a[mi].q[1] = *(const u32x4*)(p + 8);
    }
#pragma unroll
    for (int ni = 0; ni < 4; ++ni) {
      const uint16_t* p = &sB[cur][(waveN * 64 + ni * 16 + lid) * 32 + kB0];
      b[ni].q[0] = *(const u32x4*)p;
      b[ni].q[1] = *(const u32x4*)(p + 8);
    }
#pragma unroll
    for (int mi = 0; mi < 2; ++mi)
#pragma unroll
      for (int ni = 0; ni < 4; ++ni)
        acc[mi][ni] = __builtin_amdgcn_wmma_f32_16x16x32_bf16(
            false, a[mi].v, false, b[ni].v, (short)0, acc[mi][ni], false, false);

    __syncthreads();
    cur ^= 1;
  }

#pragma unroll
  for (int mi = 0; mi < 2; ++mi)
#pragma unroll
    for (int ni = 0; ni < 4; ++ni)
#pragma unroll
      for (int r = 0; r < 8; ++r) {
        int row = m0blk + waveM * 32 + mi * 16 + r + lhi * 8;
        int col = n0blk + waveN * 64 + ni * 16 + lid;
        if (row < M) {
          float v = acc[mi][ni][r];
          if (OUTF) Cf[(size_t)row * N + col] = v;
          else      Cb[(size_t)row * N + col] = f2b(v);
        }
      }
}

// ---------------------------------------------------------------------------
// Flash attention: one wave per 16-query tile, causal key chunks of 32.
// QK^T: 4 wmma, online softmax via 16-lane-half shuffles, P re-layout through
// per-wave LDS buffer, PV: 4 wmma.  V fragments are loaded early so their
// global latency overlaps the softmax VALU work.
// ---------------------------------------------------------------------------
__global__ __launch_bounds__(256) void k_attn(const uint16_t* __restrict__ Q,
                                              const uint16_t* __restrict__ Km,
                                              const uint16_t* __restrict__ Vt,
                                              uint16_t* __restrict__ O) {
  __shared__ uint16_t pbuf[8 * 16 * 32];   // 8 waves x (16x32) bf16 P tile
  const int lane = threadIdx.x & 31;
  const int wid  = threadIdx.x >> 5;
  const int lid  = lane & 15;
  const int lhi  = (lane >> 4) & 1;

  const int task  = blockIdx.x * 8 + wid;       // 0..7999 query tiles
  const int qt    = task % (SEQ / 16);          // 125 tiles per (b,h)
  const int bh    = task / (SEQ / 16);
  const int h     = bh & (NHEAD - 1);
  const int b     = bh >> 4;
  const int qbase = qt * 16;
  const int qend  = qbase + 16;
  uint16_t* pl = pbuf + wid * 512;

  // q A-fragments for head-dim chunks K=[0,32) and K=[32,64)
  Frag qa[2];
  {
    const uint16_t* qp = Q + ((size_t)(b * SEQ + qbase + lid) << 10) + (h << 6);
#pragma unroll
    for (int j = 0; j < 2; ++j) {
      const uint16_t* p = qp + j * 32 + lhi * 16;
      qa[j].q[0] = *(const u32x4*)p;
      qa[j].q[1] = *(const u32x4*)(p + 8);
    }
  }

  f32x8 o[4] = {};
  float mrow[8], lrow[8];
#pragma unroll
  for (int r = 0; r < 8; ++r) { mrow[r] = -1e30f; lrow[r] = 0.f; }

  for (int kc = 0; kc < qend; kc += 32) {
    // ---- S = q * k^T (2 key sub-tiles x 2 head-dim chunks) ----
    f32x8 s[2] = {};
#pragma unroll
    for (int ns = 0; ns < 2; ++ns) {
      int key = kc + ns * 16 + lid;
      int kcl = key < SEQ ? key : SEQ - 1;           // safe clamp, masked below
      const uint16_t* kp = Km + ((size_t)(b * SEQ + kcl) << 10) + (h << 6);
#pragma unroll
      for (int j = 0; j < 2; ++j) {
        Frag kf;
        const uint16_t* p = kp + j * 32 + lhi * 16;
        kf.q[0] = *(const u32x4*)p;
        kf.q[1] = *(const u32x4*)(p + 8);
        s[ns] = __builtin_amdgcn_wmma_f32_16x16x32_bf16(
            false, qa[j].v, false, kf.v, (short)0, s[ns], false, false);
      }
    }
    // ---- V B-fragments loaded EARLY: latency overlaps softmax below ----
    Frag vf[4];
    {
      int s0 = kc + lhi * 16;
      if (s0 > SEQ - 16) s0 = SEQ - 16;              // clamp; tail keys have P=0
#pragma unroll
      for (int ds = 0; ds < 4; ++ds) {
        int d = ds * 16 + lid;
        const uint16_t* vp = Vt + (size_t)(((b * NHEAD + h) << 6) + d) * SEQ + s0;
        vf[ds].q[0] = *(const u32x4*)vp;
        vf[ds].q[1] = *(const u32x4*)(vp + 8);
      }
    }
    // ---- scale + causal mask (C layout: lane=key, vgpr=row%8, lane>=16 -> row+8)
#pragma unroll
    for (int ns = 0; ns < 2; ++ns) {
      int key = kc + ns * 16 + lid;
#pragma unroll
      for (int r = 0; r < 8; ++r) {
        int row = qbase + r + lhi * 8;
        float v = s[ns][r] * 0.125f;                 // 1/sqrt(64)
        s[ns][r] = (key <= row && key < SEQ) ? v : -1e30f;
      }
    }
    // ---- online softmax per row ----
#pragma unroll
    for (int r = 0; r < 8; ++r) {
      float cm = fmaxf(s[0][r], s[1][r]);
#pragma unroll
      for (int mm = 1; mm < 16; mm <<= 1) cm = fmaxf(cm, __shfl_xor(cm, mm, 32));
      float mn = fmaxf(mrow[r], cm);
      float sc = __expf(mrow[r] - mn);
      mrow[r] = mn;
      float p0 = __expf(s[0][r] - mn);
      float p1 = __expf(s[1][r] - mn);
      float rs = p0 + p1;
#pragma unroll
      for (int mm = 1; mm < 16; mm <<= 1) rs += __shfl_xor(rs, mm, 32);
      lrow[r] = lrow[r] * sc + rs;
#pragma unroll
      for (int j = 0; j < 4; ++j) o[j][r] *= sc;
      s[0][r] = p0; s[1][r] = p1;
    }
    // ---- P: C layout -> A-fragment layout via per-wave LDS ----
#pragma unroll
    for (int ns = 0; ns < 2; ++ns)
#pragma unroll
      for (int r = 0; r < 8; ++r)
        pl[(r + lhi * 8) * 32 + ns * 16 + lid] = f2b(s[ns][r]);
    asm volatile("s_wait_dscnt 0" ::: "memory");
    Frag pa;
    {
      const uint16_t* p = pl + lid * 32 + lhi * 16;
      pa.q[0] = *(const u32x4*)p;
      pa.q[1] = *(const u32x4*)(p + 8);
    }
    // ---- O += P * V (4 head-dim sub-tiles) ----
#pragma unroll
    for (int ds = 0; ds < 4; ++ds)
      o[ds] = __builtin_amdgcn_wmma_f32_16x16x32_bf16(
          false, pa.v, false, vf[ds].v, (short)0, o[ds], false, false);
  }

  // ---- normalize + store bf16 in (b,s,hidden) layout for the Wo GEMM ----
#pragma unroll
  for (int ds = 0; ds < 4; ++ds)
#pragma unroll
    for (int r = 0; r < 8; ++r) {
      int row = qbase + r + lhi * 8;
      float v = o[ds][r] / lrow[r];
      O[((size_t)(b * SEQ + row) << 10) + (h << 6) + ds * 16 + lid] = f2b(v);
    }
}

// ---------------------------------------------------------------------------
extern "C" void kernel_launch(void* const* d_in, const int* in_sizes, int n_in,
                              void* d_out, int out_size, void* d_ws, size_t ws_size,
                              hipStream_t stream) {
  const float* hs  = (const float*)d_in[0];
  const float* Wq  = (const float*)d_in[1];
  const float* Wk  = (const float*)d_in[2];
  const float* Wv  = (const float*)d_in[3];
  const float* Wo  = (const float*)d_in[4];
  const float* pos = (const float*)d_in[5];

  char*  ws  = (char*)d_ws;
  size_t off = 0;
  auto alloc = [&](size_t bytes) -> void* {
    void* p = ws + off;
    off += (bytes + 255) & ~(size_t)255;
    return p;
  };
  const size_t actB = (size_t)NELEM * 2;       // bf16 activation buffer
  const size_t wB   = (size_t)HID * HID * 2;   // bf16 weight buffer
  uint16_t* hb  = (uint16_t*)alloc(actB);
  uint16_t* qb  = (uint16_t*)alloc(actB);
  uint16_t* kb  = (uint16_t*)alloc(actB);
  uint16_t* vb  = (uint16_t*)alloc(actB);
  uint16_t* vtb = (uint16_t*)alloc(actB);
  uint16_t* ab  = (uint16_t*)alloc(actB);
  uint16_t* wqb = (uint16_t*)alloc(wB);
  uint16_t* wkb = (uint16_t*)alloc(wB);
  uint16_t* wvb = (uint16_t*)alloc(wB);
  uint16_t* wob = (uint16_t*)alloc(wB);

  const int EB = 256;
  const int nW = HID * HID;

  k_addpos<<<(NELEM + EB - 1) / EB, EB, 0, stream>>>(hs, pos, hb);
  k_f2bf<<<(nW + EB - 1) / EB, EB, 0, stream>>>(Wq, wqb, nW);
  k_f2bf<<<(nW + EB - 1) / EB, EB, 0, stream>>>(Wk, wkb, nW);
  k_f2bf<<<(nW + EB - 1) / EB, EB, 0, stream>>>(Wv, wvb, nW);
  k_f2bf<<<(nW + EB - 1) / EB, EB, 0, stream>>>(Wo, wob, nW);

  dim3 ggrid(HID / 128, (MROWS + 127) / 128);  // (8, 63)
  k_gemm<false><<<ggrid, 256, 0, stream>>>(hb, wqb, nullptr, qb, MROWS, HID, HID);
  k_gemm<false><<<ggrid, 256, 0, stream>>>(hb, wkb, nullptr, kb, MROWS, HID, HID);
  k_gemm<false><<<ggrid, 256, 0, stream>>>(hb, wvb, nullptr, vb, MROWS, HID, HID);

  k_vtrans<<<(NELEM + EB - 1) / EB, EB, 0, stream>>>(vb, vtb);

  // 8000 query tiles, 8 waves/block -> 1000 blocks
  k_attn<<<(BSZ * NHEAD * (SEQ / 16)) / 8, 256, 0, stream>>>(qb, kb, vtb, ab);

  k_gemm<true><<<ggrid, 256, 0, stream>>>(ab, wob, (float*)d_out, nullptr, MROWS, HID, HID);
}
